// MultiheadAttention_17377437679652
// MI455X (gfx1250) — matrix operand order repaired and checked by reading, not compiled, not run on hardware
//
#include <hip/hip_runtime.h>
#include <hip/hip_bf16.h>
#include <math.h>

// Problem constants (from reference)
#define TT   2048
#define BBATCH 4
#define EE   512
#define HH   8
#define DD   64
#define MTOT (TT * BBATCH)            // 8192 rows for projection GEMMs
#define NHALF (BBATCH * HH * TT * DD) // 4,194,304 elements per q/k/v/ctx buffer

typedef __attribute__((ext_vector_type(16))) _Float16 v16h;
typedef __attribute__((ext_vector_type(8)))  float    v8f;

union AFrag {
    v16h v;
    _Float16 h[16];
    unsigned int u[8];
};

// A-matrix 16x32 f16, contiguous k order used here (same as B): VGPR j holds K pair starting at:
__device__ __forceinline__ int koffA(int j, int half_id) {
    return half_id * 16 + j * 2;
}
// B-matrix 32x16 f16, contiguous k order used here (same as A): VGPR j holds K pair starting at:
__device__ __forceinline__ int koffB(int j, int half_id) {
    return half_id * 16 + j * 2;
}

// ---------------------------------------------------------------------------
// f32 -> f16 conversion (weights)
// ---------------------------------------------------------------------------
__global__ void cvt_f32_f16_kernel(const float* __restrict__ src,
                                   _Float16* __restrict__ dst, int n) {
    int i = blockIdx.x * blockDim.x + threadIdx.x;
    if (i < n) dst[i] = (_Float16)src[i];
}

// ---------------------------------------------------------------------------
// Projection GEMM: Y = (X @ W^T + b) * scale, X f32 [MTOT,E], W16 f16 [E,E]
// MODE 0/1: dst = [B][H][T][D] f16 (q or k) ; MODE 2: dst = [B][H][D][T] (v^T)
// One wave computes one 16x16 output tile; k-loop over E/32 WMMA steps.
// MODE is a template parameter so the store epilogue is branch-free.
// ---------------------------------------------------------------------------
template <int MODE>
__global__ void proj_kernel(const float* __restrict__ X,
                            const _Float16* __restrict__ W16,
                            const float* __restrict__ bias,
                            _Float16* __restrict__ dst,
                            float scale) {
    const int lane    = threadIdx.x & 31;
    const int wave    = threadIdx.x >> 5;
    const int lane16  = lane & 15;
    const int half_id = lane >> 4;

    const int gw = blockIdx.x * (blockDim.x >> 5) + wave;
    const int tm = gw >> 5;   // 512 M tiles
    const int tn = gw & 31;   // 32  N tiles

    const int mrow = tm * 16 + lane16;  // A row for this lane
    const int ncol = tn * 16 + lane16;  // B column for this lane

    v8f c;
#pragma unroll
    for (int r = 0; r < 8; ++r) c[r] = 0.0f;

#pragma unroll 1
    for (int ks = 0; ks < EE / 32; ++ks) {
        AFrag a, b;
#pragma unroll
        for (int j = 0; j < 8; ++j) {
            const int kA = ks * 32 + koffA(j, half_id);
            const float2 xv = *(const float2*)(X + mrow * EE + kA);
            a.h[2 * j]     = (_Float16)xv.x;
            a.h[2 * j + 1] = (_Float16)xv.y;
            const int kB = ks * 32 + koffB(j, half_id);
            b.u[j] = *(const unsigned int*)(W16 + ncol * EE + kB);
        }
        c = __builtin_amdgcn_wmma_f32_16x16x32_f16(false, a.v, false, b.v,
                                                   (short)0, c, false, false);
    }

    const float bn = bias[ncol];
    const int h = ncol >> 6;   // head
    const int d = ncol & 63;   // dim within head
#pragma unroll
    for (int r = 0; r < 8; ++r) {
        const int m  = tm * 16 + r + 8 * half_id;
        const int t  = m >> 2;       // m / B
        const int bb = m & 3;        // m % B
        const _Float16 hv = (_Float16)((c[r] + bn) * scale);
        if (MODE == 2) {
            // v transposed: [B][H][D][T]
            dst[((bb * HH + h) * DD + d) * TT + t] = hv;
        } else {
            // q/k: [B][H][T][D]
            dst[((bb * HH + h) * TT + t) * DD + d] = hv;
        }
    }
}

// ---------------------------------------------------------------------------
// Flash attention: one wave handles 32 query rows (two 16-row tiles) of one
// (b,h), streaming all S=2048 keys in chunks of 32 with online softmax.
// K and V fragments are each reused by two WMMAs (one per query tile),
// doubling matrix-op density per byte loaded.
// ---------------------------------------------------------------------------
__global__ void __launch_bounds__(128, 1)
attn_kernel(const _Float16* __restrict__ q16,
            const _Float16* __restrict__ k16,
            const _Float16* __restrict__ vt16,
            const unsigned char* __restrict__ kpm,
            _Float16* __restrict__ ctx16) {
    __shared__ _Float16 lds[4 * 2 * 16 * 32];   // per wave: 2 x (16x32) P stage

    const int lane    = threadIdx.x & 31;
    const int wave    = threadIdx.x >> 5;
    const int lane16  = lane & 15;
    const int half_id = lane >> 4;

    const int gw = blockIdx.x * (blockDim.x >> 5) + wave;
    const int bh = gw >> 6;    // b*H + h  (32)
    const int qg = gw & 63;    // 32-row query group (T/32)
    const int bb = bh >> 3;
    const int hh = bh & 7;

    const _Float16* Qb = q16 + (bh * TT + qg * 32) * DD;
    const _Float16* Kb = k16 + bh * TT * DD;
    const _Float16* Vb = vt16 + bh * DD * TT;
    const unsigned char* mk = kpm + bb * TT;

    // Persistent Q A-fragments: [query tile][K-step]
    AFrag qa[2][2];
#pragma unroll
    for (int qt = 0; qt < 2; ++qt)
#pragma unroll
        for (int s = 0; s < 2; ++s)
#pragma unroll
            for (int j = 0; j < 8; ++j) {
                const int kA = s * 32 + koffA(j, half_id);
                qa[qt][s].u[j] =
                    *(const unsigned int*)(Qb + (qt * 16 + lane16) * DD + kA);
            }

    v8f oacc[2][4];
    float mrow[2][8], lrow[2][8];
#pragma unroll
    for (int qt = 0; qt < 2; ++qt) {
#pragma unroll
        for (int blk = 0; blk < 4; ++blk)
#pragma unroll
            for (int r = 0; r < 8; ++r) oacc[qt][blk][r] = 0.0f;
#pragma unroll
        for (int r = 0; r < 8; ++r) { mrow[qt][r] = -INFINITY; lrow[qt][r] = 0.0f; }
    }

#pragma unroll 1
    for (int sc = 0; sc < TT / 32; ++sc) {
        const int sbase = sc * 32;

        // prefetch next chunk's K rows and V columns into cache
        if (sc + 1 < TT / 32) {
            __builtin_prefetch(Kb + (sbase + 32 + lane) * DD, 0, 3);
            __builtin_prefetch(Vb + lane * TT + sbase + 32, 0, 3);
            __builtin_prefetch(Vb + (32 + lane) * TT + sbase + 32, 0, 3);
        }

        // ---- scores: [query tile][key tile], K=64 contraction in 2 steps ----
        v8f sf[2][2];
#pragma unroll
        for (int qt = 0; qt < 2; ++qt)
#pragma unroll
            for (int tile = 0; tile < 2; ++tile)
#pragma unroll
                for (int r = 0; r < 8; ++r) sf[qt][tile][r] = 0.0f;

#pragma unroll
        for (int tile = 0; tile < 2; ++tile) {
#pragma unroll
            for (int step = 0; step < 2; ++step) {
                AFrag kb;  // B = K^T: col = key s, pairs along head-dim
#pragma unroll
                for (int j = 0; j < 8; ++j) {
                    const int kB = step * 32 + koffB(j, half_id);
                    kb.u[j] = *(const unsigned int*)(
                        Kb + (sbase + tile * 16 + lane16) * DD + kB);
                }
                // reuse K fragment for both query tiles
#pragma unroll
                for (int qt = 0; qt < 2; ++qt)
                    sf[qt][tile] = __builtin_amdgcn_wmma_f32_16x16x32_f16(
                        false, qa[qt][step].v, false, kb.v, (short)0,
                        sf[qt][tile], false, false);
            }
        }

        // key padding mask (column-wise, identical for both query tiles)
        const bool mv0 = mk[sbase + lane16] != 0;
        const bool mv1 = mk[sbase + 16 + lane16] != 0;
#pragma unroll
        for (int qt = 0; qt < 2; ++qt) {
            if (mv0) {
#pragma unroll
                for (int r = 0; r < 8; ++r) sf[qt][0][r] = -INFINITY;
            }
            if (mv1) {
#pragma unroll
                for (int r = 0; r < 8; ++r) sf[qt][1][r] = -INFINITY;
            }
        }

        // ---- online softmax + P staging, per query tile ----
        AFrag pa[2];
#pragma unroll
        for (int qt = 0; qt < 2; ++qt) {
            _Float16* pbuf = lds + (wave * 2 + qt) * 16 * 32;
#pragma unroll
            for (int r = 0; r < 8; ++r) {
                float mx = fmaxf(sf[qt][0][r], sf[qt][1][r]);
                mx = fmaxf(mx, __shfl_xor(mx, 1, 32));
                mx = fmaxf(mx, __shfl_xor(mx, 2, 32));
                mx = fmaxf(mx, __shfl_xor(mx, 4, 32));
                mx = fmaxf(mx, __shfl_xor(mx, 8, 32));
                const float mnew  = fmaxf(mrow[qt][r], mx);
                const float alpha = __expf(mrow[qt][r] - mnew);
                const float p0 = __expf(sf[qt][0][r] - mnew);
                const float p1 = __expf(sf[qt][1][r] - mnew);
                float rs = p0 + p1;
                rs += __shfl_xor(rs, 1, 32);
                rs += __shfl_xor(rs, 2, 32);
                rs += __shfl_xor(rs, 4, 32);
                rs += __shfl_xor(rs, 8, 32);
                lrow[qt][r] = lrow[qt][r] * alpha + rs;
                mrow[qt][r] = mnew;
                oacc[qt][0][r] *= alpha; oacc[qt][1][r] *= alpha;
                oacc[qt][2][r] *= alpha; oacc[qt][3][r] *= alpha;
                // stage P (f16): C-layout -> A-layout transpose via LDS
                pbuf[(r + 8 * half_id) * 32 + lane16]      = (_Float16)p0;
                pbuf[(r + 8 * half_id) * 32 + 16 + lane16] = (_Float16)p1;
            }
#pragma unroll
            for (int j = 0; j < 8; ++j) {
                const int kA = koffA(j, half_id);
                pa[qt].u[j] = *(const unsigned int*)(pbuf + lane16 * 32 + kA);
            }
        }

        // ---- P (16x32) x V (32x64): V fragment reused by both query tiles ----
#pragma unroll
        for (int blk = 0; blk < 4; ++blk) {
            AFrag vb;  // B = V chunk: col = d, pairs along s (contiguous in v^T)
#pragma unroll
            for (int j = 0; j < 8; ++j) {
                const int kB = koffB(j, half_id);
                vb.u[j] = *(const unsigned int*)(
                    Vb + (blk * 16 + lane16) * TT + sbase + kB);
            }
#pragma unroll
            for (int qt = 0; qt < 2; ++qt)
                oacc[qt][blk] = __builtin_amdgcn_wmma_f32_16x16x32_f16(
                    false, pa[qt].v, false, vb.v, (short)0, oacc[qt][blk],
                    false, false);
        }
    }

    // ---- normalize and store ctx as f16 [T*B, E] row-major ----
#pragma unroll
    for (int qt = 0; qt < 2; ++qt)
#pragma unroll
        for (int r = 0; r < 8; ++r) {
            const float inv = 1.0f / lrow[qt][r];
            const int t = qg * 32 + qt * 16 + r + 8 * half_id;
#pragma unroll
            for (int blk = 0; blk < 4; ++blk) {
                const int d = blk * 16 + lane16;
                ctx16[(t * BBATCH + bb) * EE + hh * DD + d] =
                    (_Float16)(oacc[qt][blk][r] * inv);
            }
        }
}

// ---------------------------------------------------------------------------
// Output projection: out = ctx16 @ Wo^T + bo, f16 x f16 -> f32
// ---------------------------------------------------------------------------
__global__ void outproj_kernel(const _Float16* __restrict__ A16,
                               const _Float16* __restrict__ W16,
                               const float* __restrict__ bias,
                               float* __restrict__ out) {
    const int lane    = threadIdx.x & 31;
    const int wave    = threadIdx.x >> 5;
    const int lane16  = lane & 15;
    const int half_id = lane >> 4;

    const int gw = blockIdx.x * (blockDim.x >> 5) + wave;
    const int tm = gw >> 5;
    const int tn = gw & 31;

    const int mrow = tm * 16 + lane16;
    const int ncol = tn * 16 + lane16;

    v8f c;
#pragma unroll
    for (int r = 0; r < 8; ++r) c[r] = 0.0f;

#pragma unroll 1
    for (int ks = 0; ks < EE / 32; ++ks) {
        AFrag a, b;
#pragma unroll
        for (int j = 0; j < 8; ++j) {
            const int kA = ks * 32 + koffA(j, half_id);
            a.u[j] = *(const unsigned int*)(A16 + mrow * EE + kA);
            const int kB = ks * 32 + koffB(j, half_id);
            b.u[j] = *(const unsigned int*)(W16 + ncol * EE + kB);
        }
        c = __builtin_amdgcn_wmma_f32_16x16x32_f16(false, a.v, false, b.v,
                                                   (short)0, c, false, false);
    }

    const float bn = bias[ncol];
#pragma unroll
    for (int r = 0; r < 8; ++r) {
        const int m = tm * 16 + r + 8 * half_id;
        out[m * EE + ncol] = c[r] + bn;
    }
}

// ---------------------------------------------------------------------------
extern "C" void kernel_launch(void* const* d_in, const int* in_sizes, int n_in,
                              void* d_out, int out_size, void* d_ws, size_t ws_size,
                              hipStream_t stream) {
    const float* query = (const float*)d_in[0];
    const float* key   = (const float*)d_in[1];
    const float* value = (const float*)d_in[2];
    const unsigned char* kpm = (const unsigned char*)d_in[3];
    const float* Wq = (const float*)d_in[4];
    const float* bq = (const float*)d_in[5];
    const float* Wk = (const float*)d_in[6];
    const float* bk = (const float*)d_in[7];
    const float* Wv = (const float*)d_in[8];
    const float* bv = (const float*)d_in[9];
    const float* Wo = (const float*)d_in[10];
    const float* bo = (const float*)d_in[11];
    float* out = (float*)d_out;

    // workspace layout (f16): 4 weight matrices, q, k, v^T, ctx (~34 MB)
    _Float16* ws   = (_Float16*)d_ws;
    _Float16* W16  = ws;
    _Float16* q16  = W16 + 4 * EE * EE;
    _Float16* k16  = q16 + NHALF;
    _Float16* vt16 = k16 + NHALF;
    _Float16* c16  = vt16 + NHALF;

    const int nW = EE * EE;
    const int cvtBlocks = (nW + 255) / 256;
    cvt_f32_f16_kernel<<<cvtBlocks, 256, 0, stream>>>(Wq, W16 + 0 * nW, nW);
    cvt_f32_f16_kernel<<<cvtBlocks, 256, 0, stream>>>(Wk, W16 + 1 * nW, nW);
    cvt_f32_f16_kernel<<<cvtBlocks, 256, 0, stream>>>(Wv, W16 + 2 * nW, nW);
    cvt_f32_f16_kernel<<<cvtBlocks, 256, 0, stream>>>(Wo, W16 + 3 * nW, nW);

    const float scaling = 0.00125f;  // (D^-0.5)/100 = 0.125/100
    dim3 blk(128);                   // 4 waves / block
    dim3 gridProj((MTOT / 16) * (EE / 16) / 4);  // 4096 blocks
    proj_kernel<0><<<gridProj, blk, 0, stream>>>(query, W16 + 0 * nW, bq, q16, scaling);
    proj_kernel<1><<<gridProj, blk, 0, stream>>>(key,   W16 + 1 * nW, bk, k16, 1.0f);
    proj_kernel<2><<<gridProj, blk, 0, stream>>>(value, W16 + 2 * nW, bv, vt16, 1.0f);

    // 32 (b,h) x 64 query groups = 2048 waves, 4 waves/block
    dim3 gridAttn((BBATCH * HH) * (TT / 32) / 4);  // 512 blocks
    attn_kernel<<<gridAttn, blk, 0, stream>>>(q16, k16, vt16, kpm, c16);

    outproj_kernel<<<gridProj, blk, 0, stream>>>(c16, W16 + 3 * nW, bo, out);
}
